// GraspMovNet_42640435315376
// MI455X (gfx1250) — matrix operand order repaired and checked by reading, not compiled, not run on hardware
//
#include <hip/hip_runtime.h>

// ---------------------------------------------------------------------------
// CDNA5 (gfx1250) transformer encoder forward.
// GEMMs: v_wmma_f32_16x16x32_bf16 (wave32), double-buffered LDS tiles fed by
// GLOBAL_LOAD_ASYNC_TO_LDS_B128 (ASYNCcnt) when the toolchain exposes it.
// ---------------------------------------------------------------------------

typedef __attribute__((ext_vector_type(16))) __bf16 v16bf;
typedef __attribute__((ext_vector_type(8)))  float  v8f;
typedef __attribute__((ext_vector_type(4)))  unsigned int u32x4;

union BFrag16 { v16bf v; u32x4 u[2]; };

#define TOK    16384      // 16 * 1024 token rows
#define DMODEL 512
#define DFF_   2048
#define NLAYER 4

#define BM 128
#define BN 128
#define BK 32

#if defined(__AMDGCN__) && __has_builtin(__builtin_amdgcn_global_load_async_to_lds_b128) && __has_builtin(__builtin_amdgcn_s_wait_asynccnt)
#define USE_ASYNC_COPY 1
#else
#define USE_ASYNC_COPY 0
#endif

#if USE_ASYNC_COPY
// Builtin prototype (from hipcc diagnostic): param0 = global (addrspace(1))
// pointer to 'int __attribute__((vector_size(16)))', param1 = LDS
// (addrspace(3)) pointer to same, then imm offset, imm cpol.
typedef int v4i_vs __attribute__((vector_size(16)));
typedef __attribute__((address_space(1))) v4i_vs* gas_v4i;
typedef __attribute__((address_space(3))) v4i_vs* las_v4i;

__device__ __forceinline__ void async_ld16(const __bf16* g, __bf16* l) {
  __builtin_amdgcn_global_load_async_to_lds_b128(
      (gas_v4i)(g), (las_v4i)(l), 0, 0);
}
#endif

// ---------------- weight transpose + f32 -> bf16 (Bt = B^T, N x K) ---------
__global__ void k_transpose_bf16(const float* __restrict__ src,
                                 __bf16* __restrict__ dst, int K, int N) {
  int total = K * N;
  for (int i = blockIdx.x * blockDim.x + threadIdx.x; i < total;
       i += gridDim.x * blockDim.x) {
    int k = i / N, n = i - k * N;
    dst[(size_t)n * K + k] = (__bf16)src[i];
  }
}

// ---------------- embedding + sinusoidal positional encoding ---------------
__global__ __launch_bounds__(256) void k_embed(
    const float* __restrict__ x, const int* __restrict__ pad,
    const float* __restrict__ encW, const float* __restrict__ encb,
    float* __restrict__ h, __bf16* __restrict__ hb) {
  __shared__ float xs[16];
  int row = blockIdx.x;            // n*1024 + t
  int t   = row & 1023;
  int tid = threadIdx.x;
  if (tid < 16) xs[tid] = x[row * 16 + tid];
  __syncthreads();
  int pm = pad[row];
  for (int d = tid; d < DMODEL; d += 256) {
    float s = encb[d];
#pragma unroll
    for (int i = 0; i < 16; ++i) s += xs[i] * encW[i * DMODEL + d];
    s *= 22.627416997969522f;      // sqrt(512)
    if (pm != 0) {
      // base = t / 10000^(d/512)
      float base = (float)t * expf((float)d * (-9.210340371976184f / (float)DMODEL));
      s += (d & 1) ? cosf(base) : sinf(base);
    }
    size_t idx = (size_t)row * DMODEL + d;
    h[idx]  = s;
    hb[idx] = (__bf16)s;
  }
}

// ---------------- bf16 WMMA GEMM: C(MxN,f32) = A(MxK) * Bt(NxK)^T ----------
// 128x128 block tile, BK=32, 8 waves in 4(M)x2(N), each wave 32x64 = 2x4
// 16x16 WMMA tiles. Double-buffered LDS (32 KB) with async global->LDS copy.
__global__ __launch_bounds__(256) void k_gemm_bf16(
    const __bf16* __restrict__ A, const __bf16* __restrict__ Bt,
    float* __restrict__ C, int M, int N, int K) {
  __shared__ __align__(16) __bf16 sA[2][BM][BK];   // 2 x 8 KB
  __shared__ __align__(16) __bf16 sB[2][BN][BK];   // 2 x 8 KB
  const int tid  = threadIdx.x;
  const int lane = tid & 31;
  const int wave = tid >> 5;           // 8 waves
  const int wm   = wave >> 1;          // 0..3 -> 32-row strip
  const int wn   = wave & 1;           // 0..1 -> 64-col strip
  const int half = lane >> 4;          // lane group 0/1
  const int l15  = lane & 15;
  const size_t blockM = (size_t)blockIdx.x * BM;
  const size_t blockN = (size_t)blockIdx.y * BN;

  // per-thread 16B chunk slots: chunks tid and tid+256 of 512 (128 rows x 4)
  const int r0 = tid >> 2;             // 0..63
  const int c0 = (tid & 3) << 3;       // 0,8,16,24
  const int r1 = r0 + 64;

  v8f acc[2][4] = {};

  const int nk = K / BK;

  auto load_tile = [&](int k0, int buf) {
    const __bf16* a0 = &A[(blockM + r0) * (size_t)K + k0 + c0];
    const __bf16* a1 = &A[(blockM + r1) * (size_t)K + k0 + c0];
    const __bf16* b0 = &Bt[(blockN + r0) * (size_t)K + k0 + c0];
    const __bf16* b1 = &Bt[(blockN + r1) * (size_t)K + k0 + c0];
#if USE_ASYNC_COPY
    async_ld16(a0, &sA[buf][r0][c0]);
    async_ld16(a1, &sA[buf][r1][c0]);
    async_ld16(b0, &sB[buf][r0][c0]);
    async_ld16(b1, &sB[buf][r1][c0]);
#else
    *(u32x4*)&sA[buf][r0][c0] = *(const u32x4*)a0;
    *(u32x4*)&sA[buf][r1][c0] = *(const u32x4*)a1;
    *(u32x4*)&sB[buf][r0][c0] = *(const u32x4*)b0;
    *(u32x4*)&sB[buf][r1][c0] = *(const u32x4*)b1;
#endif
  };

  load_tile(0, 0);

  for (int i = 0; i < nk; ++i) {
    const int buf = i & 1;
    // prefetch next tile into the other buffer (safe: its last compute was
    // fenced by the trailing __syncthreads of iteration i-1)
    if (i + 1 < nk) {
      load_tile((i + 1) * BK, buf ^ 1);
#if USE_ASYNC_COPY
      __builtin_amdgcn_s_wait_asynccnt(4);   // tile i's 4 loads complete
#endif
    } else {
#if USE_ASYNC_COPY
      __builtin_amdgcn_s_wait_asynccnt(0);
#endif
    }
    __syncthreads();                          // tile i visible to all waves

    BFrag16 fa[2], fb[4];
#pragma unroll
    for (int tm = 0; tm < 2; ++tm) {
      // A 16x32 layout: lanes 0-15 rows M, K = {half*8+0..7, 16+half*8+0..7}
      int ar = wm * 32 + tm * 16 + l15;
      fa[tm].u[0] = *(const u32x4*)&sA[buf][ar][half * 8];
      fa[tm].u[1] = *(const u32x4*)&sA[buf][ar][16 + half * 8];
    }
#pragma unroll
    for (int tn = 0; tn < 4; ++tn) {
      // B 32x16 layout: lane = N col, K = {half*8+0..7, 16+half*8+0..7}
      int br = wn * 64 + tn * 16 + l15;
      fb[tn].u[0] = *(const u32x4*)&sB[buf][br][half * 8];
      fb[tn].u[1] = *(const u32x4*)&sB[buf][br][16 + half * 8];
    }
#pragma unroll
    for (int tm = 0; tm < 2; ++tm)
#pragma unroll
      for (int tn = 0; tn < 4; ++tn)
        acc[tm][tn] = __builtin_amdgcn_wmma_f32_16x16x32_bf16(
            false, fa[tm].v, false, fb[tn].v, (short)0, acc[tm][tn],
            false, false);
    __syncthreads();                          // protect buf before reuse
  }

  // C/D layout: lane = N col (mod 16), VGPR r -> row half*8 + r
#pragma unroll
  for (int tm = 0; tm < 2; ++tm) {
#pragma unroll
    for (int tn = 0; tn < 4; ++tn) {
      size_t rr = blockM + wm * 32 + tm * 16 + half * 8;
      size_t cc = blockN + wn * 64 + tn * 16 + l15;
#pragma unroll
      for (int r = 0; r < 8; ++r)
        C[(rr + r) * N + cc] = acc[tm][tn][r];
    }
  }
}

// ---------------- per-token attention (source's 64x64-over-heads quirk) ----
__global__ __launch_bounds__(256) void k_attention(
    const float* __restrict__ q, const float* __restrict__ k,
    const float* __restrict__ v, const int* __restrict__ pad,
    __bf16* __restrict__ out) {
  __shared__ float qs[DMODEL], ks[DMODEL], vs[DMODEL];
  __shared__ float e[64][65];
  __shared__ float rmax[64][4], rsum[64][4];
  int row = blockIdx.x;
  int tid = threadIdx.x;
  size_t base = (size_t)row * DMODEL;
  for (int i = tid; i < DMODEL; i += 256) {
    qs[i] = q[base + i]; ks[i] = k[base + i]; vs[i] = v[base + i];
  }
  __syncthreads();
  int pm = pad[row];
  int a = tid >> 2, g = tid & 3;
  int b0 = g * 16;
  // energy[a][b] = sum_h Qh[a,h]*Kh[h,b] / 32 ; masked rows -> all equal
  float lmax = -3.4e38f;
  for (int b = b0; b < b0 + 16; ++b) {
    float s = 0.f;
#pragma unroll
    for (int hh = 0; hh < 8; ++hh) s += qs[hh * 64 + a] * ks[hh * 64 + b];
    s = pm ? s * (1.0f / 32.0f) : 0.0f;
    e[a][b] = s;
    lmax = fmaxf(lmax, s);
  }
  rmax[a][g] = lmax;
  __syncthreads();
  float m = fmaxf(fmaxf(rmax[a][0], rmax[a][1]), fmaxf(rmax[a][2], rmax[a][3]));
  float lsum = 0.f;
  for (int b = b0; b < b0 + 16; ++b) {
    float t = expf(e[a][b] - m);
    e[a][b] = t;
    lsum += t;
  }
  rsum[a][g] = lsum;
  __syncthreads();
  float inv = 1.0f / (rsum[a][0] + rsum[a][1] + rsum[a][2] + rsum[a][3]);
  for (int b = b0; b < b0 + 16; ++b) e[a][b] *= inv;
  __syncthreads();
  // head[a,hh] -> out col a*8+hh ; Vh[b,hh] = v[hh*64+b]
  for (int o = tid; o < DMODEL; o += 256) {
    int aa = o >> 3, hh = o & 7;
    float s = 0.f;
#pragma unroll
    for (int b = 0; b < 64; ++b) s += e[aa][b] * vs[hh * 64 + b];
    out[base + o] = (__bf16)s;
  }
}

// ---------------- residual + (bias) + layernorm ----------------------------
__global__ __launch_bounds__(256) void k_add_ln(
    const float* __restrict__ src, const float* __restrict__ bias,
    const float* __restrict__ resid, const float* __restrict__ g,
    const float* __restrict__ b, float* __restrict__ h,
    __bf16* __restrict__ hb) {
  __shared__ float rs[256], rq[256];
  int row = blockIdx.x, tid = threadIdx.x;
  size_t base = (size_t)row * DMODEL;
  int d0 = tid, d1 = tid + 256;
  float x0 = src[base + d0] + resid[base + d0];
  float x1 = src[base + d1] + resid[base + d1];
  if (bias) { x0 += bias[d0]; x1 += bias[d1]; }
  rs[tid] = x0 + x1;
  rq[tid] = x0 * x0 + x1 * x1;
  __syncthreads();
  for (int s = 128; s > 0; s >>= 1) {
    if (tid < s) { rs[tid] += rs[tid + s]; rq[tid] += rq[tid + s]; }
    __syncthreads();
  }
  float mean = rs[0] * (1.0f / 512.0f);
  float var  = rq[0] * (1.0f / 512.0f) - mean * mean;
  float rstd = rsqrtf(var + 1e-5f);
  float y0 = (x0 - mean) * rstd * g[d0] + b[d0];
  float y1 = (x1 - mean) * rstd * g[d1] + b[d1];
  h[base + d0] = y0;           h[base + d1] = y1;
  hb[base + d0] = (__bf16)y0;  hb[base + d1] = (__bf16)y1;
}

// ---------------- bias + relu -> bf16 --------------------------------------
__global__ void k_bias_relu(const float* __restrict__ src,
                            const float* __restrict__ bias,
                            __bf16* __restrict__ dst, long long total, int cols) {
  for (long long i = blockIdx.x * (long long)blockDim.x + threadIdx.x; i < total;
       i += (long long)gridDim.x * blockDim.x) {
    float vv = src[i] + bias[(int)(i % cols)];
    dst[i] = (__bf16)fmaxf(vv, 0.0f);
  }
}

// ---------------- final projection: (16, 524288) @ (524288, 3) -------------
__global__ __launch_bounds__(256) void k_final_partial(
    const float* __restrict__ h, const float* __restrict__ outW,
    float* __restrict__ partial) {
  __shared__ float rs[256];
  int chunk = blockIdx.x;           // 0..63
  int nj    = blockIdx.y;           // 0..47 = n*3 + j
  int n = nj / 3, j = nj - n * 3;
  int tid = threadIdx.x;
  const float* hr = h + (size_t)n * (DMODEL * 1024);
  float s = 0.f;
  int i0 = chunk * 8192;
  for (int i = i0 + tid; i < i0 + 8192; i += 256)
    s += hr[i] * outW[(size_t)i * 3 + j];
  rs[tid] = s;
  __syncthreads();
  for (int st = 128; st > 0; st >>= 1) {
    if (tid < st) rs[tid] += rs[tid + st];
    __syncthreads();
  }
  if (tid == 0) partial[nj * 64 + chunk] = rs[0];
}

__global__ void k_final_reduce(const float* __restrict__ partial,
                               const float* __restrict__ outb,
                               float* __restrict__ out) {
  __shared__ float rs[64];
  int nj = blockIdx.x, tid = threadIdx.x;   // 64 threads
  rs[tid] = partial[nj * 64 + tid];
  __syncthreads();
  for (int st = 32; st > 0; st >>= 1) {
    if (tid < st) rs[tid] += rs[tid + st];
    __syncthreads();
  }
  if (tid == 0) out[nj] = rs[0] + outb[nj % 3];
}

// ---------------------------------------------------------------------------
extern "C" void kernel_launch(void* const* d_in, const int* in_sizes, int n_in,
                              void* d_out, int out_size, void* d_ws, size_t ws_size,
                              hipStream_t stream) {
  const float* x    = (const float*)d_in[0];
  const int*   pad  = (const int*)  d_in[1];
  const float* encW = (const float*)d_in[2];
  const float* encb = (const float*)d_in[3];
  const float* Wq   = (const float*)d_in[4];
  const float* Wk   = (const float*)d_in[5];
  const float* Wv   = (const float*)d_in[6];
  const float* Wo   = (const float*)d_in[7];
  const float* ffW1 = (const float*)d_in[8];
  const float* ffb1 = (const float*)d_in[9];
  const float* ffW2 = (const float*)d_in[10];
  const float* ffb2 = (const float*)d_in[11];
  const float* ln1g = (const float*)d_in[12];
  const float* ln1b = (const float*)d_in[13];
  const float* ln2g = (const float*)d_in[14];
  const float* ln2b = (const float*)d_in[15];
  const float* outW = (const float*)d_in[16];
  const float* outb = (const float*)d_in[17];
  float* out = (float*)d_out;

  // ---- workspace sub-allocation (256B aligned) ----
  char* base = (char*)d_ws;
  size_t off = 0;
  auto alloc = [&](size_t bytes) -> void* {
    void* p = base + off;
    off = (off + bytes + 255) & ~(size_t)255;
    return p;
  };
  float*  h    = (float*) alloc((size_t)TOK * DMODEL * 4);
  __bf16* hb   = (__bf16*)alloc((size_t)TOK * DMODEL * 2);
  float*  qb   = (float*) alloc((size_t)TOK * DMODEL * 4);
  float*  kb   = (float*) alloc((size_t)TOK * DMODEL * 4);
  float*  vb   = (float*) alloc((size_t)TOK * DMODEL * 4);
  __bf16* ab   = (__bf16*)alloc((size_t)TOK * DMODEL * 2);
  float*  tmpD = (float*) alloc((size_t)TOK * DMODEL * 4);
  float*  f1   = (float*) alloc((size_t)TOK * DFF_ * 4);
  __bf16* f1b  = (__bf16*)alloc((size_t)TOK * DFF_ * 2);
  const size_t WPL = 3145728;   // bf16 elements of transposed weights per layer
  __bf16* wT   = (__bf16*)alloc((size_t)NLAYER * WPL * 2);
  float*  part = (float*) alloc(48 * 64 * 4);
  (void)ws_size; (void)in_sizes; (void)n_in; (void)out_size;

  // ---- transpose + cast all weights to bf16 B^T layout ----
  for (int l = 0; l < NLAYER; ++l) {
    __bf16* wl = wT + (size_t)l * WPL;
    size_t dd = (size_t)l * DMODEL * DMODEL;
    k_transpose_bf16<<<512, 256, 0, stream>>>(Wq + dd, wl + 0,      DMODEL, DMODEL);
    k_transpose_bf16<<<512, 256, 0, stream>>>(Wk + dd, wl + 262144, DMODEL, DMODEL);
    k_transpose_bf16<<<512, 256, 0, stream>>>(Wv + dd, wl + 524288, DMODEL, DMODEL);
    k_transpose_bf16<<<512, 256, 0, stream>>>(Wo + dd, wl + 786432, DMODEL, DMODEL);
    k_transpose_bf16<<<2048, 256, 0, stream>>>(ffW1 + (size_t)l * DMODEL * DFF_,
                                               wl + 1048576, DMODEL, DFF_);
    k_transpose_bf16<<<2048, 256, 0, stream>>>(ffW2 + (size_t)l * DFF_ * DMODEL,
                                               wl + 2097152, DFF_, DMODEL);
  }

  // ---- input encoding + positional encoding ----
  k_embed<<<TOK, 256, 0, stream>>>(x, pad, encW, encb, h, hb);

  dim3 gD(TOK / BM, DMODEL / BN);   // (128, 4)
  dim3 gF(TOK / BM, DFF_ / BN);     // (128, 16)

  for (int l = 0; l < NLAYER; ++l) {
    __bf16* wl = wT + (size_t)l * WPL;
    // Q, K, V projections
    k_gemm_bf16<<<gD, 256, 0, stream>>>(hb, wl + 0,      qb, TOK, DMODEL, DMODEL);
    k_gemm_bf16<<<gD, 256, 0, stream>>>(hb, wl + 262144, kb, TOK, DMODEL, DMODEL);
    k_gemm_bf16<<<gD, 256, 0, stream>>>(hb, wl + 524288, vb, TOK, DMODEL, DMODEL);
    // per-token attention
    k_attention<<<TOK, 256, 0, stream>>>(qb, kb, vb, pad, ab);
    // output projection
    k_gemm_bf16<<<gD, 256, 0, stream>>>(ab, wl + 786432, tmpD, TOK, DMODEL, DMODEL);
    // residual + LN1
    k_add_ln<<<TOK, 256, 0, stream>>>(tmpD, (const float*)nullptr, h,
                                      ln1g + l * DMODEL, ln1b + l * DMODEL, h, hb);
    // FFN
    k_gemm_bf16<<<gF, 256, 0, stream>>>(hb, wl + 1048576, f1, TOK, DFF_, DMODEL);
    k_bias_relu<<<4096, 256, 0, stream>>>(f1, ffb1 + (size_t)l * DFF_, f1b,
                                          (long long)TOK * DFF_, DFF_);
    k_gemm_bf16<<<gD, 256, 0, stream>>>(f1b, wl + 2097152, tmpD, TOK, DMODEL, DFF_);
    // bias + residual + LN2
    k_add_ln<<<TOK, 256, 0, stream>>>(tmpD, ffb2 + (size_t)l * DMODEL, h,
                                      ln2g + l * DMODEL, ln2b + l * DMODEL, h, hb);
  }

  // ---- final flatten-project (deterministic two-stage reduction) ----
  dim3 gP(64, 48);
  k_final_partial<<<gP, 256, 0, stream>>>(h, outW, part);
  k_final_reduce<<<48, 64, 0, stream>>>(part, outb, out);
}
